// EncoderBlock_71262097375626
// MI455X (gfx1250) — matrix operand order repaired and checked by reading, not compiled, not run on hardware
//
#include <hip/hip_runtime.h>
#include <hip/hip_bf16.h>

typedef _Float16 f16;
typedef __attribute__((ext_vector_type(16))) _Float16 v16h;
typedef __attribute__((ext_vector_type(8)))  float    v8f;
typedef __attribute__((ext_vector_type(4)))  unsigned u32x4;

#define DEV __device__ __forceinline__

static constexpr int B_ = 4, S_ = 2048, D_ = 1024, H_ = 16, DK_ = 64, DFF_ = 4096;
static constexpr int M_ = B_ * S_;  // 8192 rows

// ---------------------------------------------------------------- WMMA helpers

DEV v8f wmma_f16(v16h a, v16h b, v8f c) {
  // D = A(16x32 f16) * B(32x16 f16) + C(16x16 f32)
  return __builtin_amdgcn_wmma_f32_16x16x32_f16(false, a, false, b, (short)0, c,
                                                false, false);
}

// A fragment 16x32 f16, row-major source (ld halves). Per ISA layout the 16
// halves of a lane are two contiguous 8-half runs: [hi*8, hi*8+7] and +16.
DEV v16h load_a(const f16* p, int ld, int k0, int lane) {
  const int hi = (lane >> 4) & 1;
  const int m  = lane & 15;
  union { v16h v; u32x4 q[2]; } r;
  const f16* row = p + (size_t)m * ld + k0 + hi * 8;
  r.q[0] = *(const u32x4*)(row);
  r.q[1] = *(const u32x4*)(row + 16);
  return r.v;
}

// B fragment 32x16 f16 where B[k][n] = p[n*ld + k0 + k] ("A @ W^T" weights).
// Mirrored layout: lane n holds two 8-half runs [hi*8, hi*8+7] and +16.
DEV v16h load_bt(const f16* p, int ld, int k0, int lane) {
  const int hi = (lane >> 4) & 1;
  const int n  = lane & 15;
  union { v16h v; u32x4 q[2]; } r;
  const f16* row = p + (size_t)n * ld + k0 + hi * 8;
  r.q[0] = *(const u32x4*)(row);
  r.q[1] = *(const u32x4*)(row + 16);
  return r.v;
}

// gfx1250 async copy: global -> LDS, 16 bytes per lane, tracked by ASYNCcnt.
// Low 32 bits of a generic LDS pointer are the wave-relative LDS byte offset.
DEV void async_copy_b128(const f16* gsrc, const f16* lds_dst) {
  unsigned dst = (unsigned)(size_t)lds_dst;
  asm volatile("global_load_async_to_lds_b128 %0, %1, off"
               :: "v"(dst), "v"(gsrc) : "memory");
}
DEV void wait_async0() { asm volatile("s_wait_asynccnt 0x0" ::: "memory"); }

// ---------------------------------------------------------------- small kernels

__global__ __launch_bounds__(256) void f32_to_f16_kernel(const float* __restrict__ in,
                                                         f16* __restrict__ out, int n) {
  int i = blockIdx.x * 256 + threadIdx.x;
  if (i < n) out[i] = (f16)in[i];
}

// LayerNorm per row of D_=1024, torch-style: a*(x-mean)/(std_unbiased+eps)+c.
__global__ __launch_bounds__(256) void ln_kernel(const float* __restrict__ x,
                                                 const float* __restrict__ alpha,
                                                 const float* __restrict__ beta,
                                                 f16* __restrict__ out) {
  __shared__ float rs[256], rq[256];
  const int tid = threadIdx.x;
  const int row = blockIdx.x;
  const float* xr = x + (size_t)row * D_;
  float v[4], s = 0.f, sq = 0.f;
#pragma unroll
  for (int i = 0; i < 4; ++i) {
    v[i] = xr[tid + i * 256];
    s += v[i];
    sq += v[i] * v[i];
  }
  rs[tid] = s; rq[tid] = sq;
  __syncthreads();
  for (int o = 128; o > 0; o >>= 1) {
    if (tid < o) { rs[tid] += rs[tid + o]; rq[tid] += rq[tid + o]; }
    __syncthreads();
  }
  const float mean = rs[0] * (1.0f / D_);
  const float var  = (rq[0] - (float)D_ * mean * mean) * (1.0f / (D_ - 1));
  const float inv  = alpha[0] / (sqrtf(var) + 1e-6f);
  const float bta  = beta[0];
#pragma unroll
  for (int i = 0; i < 4; ++i)
    out[(size_t)row * D_ + tid + i * 256] = (f16)((v[i] - mean) * inv + bta);
}

// ---------------------------------------------------------------- GEMM (Y = A @ W^T + b)
// Block = 8 waves = 128 rows x 64 cols. A 64x64 weight slab (two K-steps) is
// staged into LDS via async-to-LDS (double buffered) and shared by all 8 waves:
// one barrier per 8 WMMAs/wave, 8x cut in L2 weight traffic.

enum { EPI_F16 = 0, EPI_F16_HEADS = 1, EPI_F16_RELU = 2, EPI_F32_RES = 3,
       EPI_F16_HEADS_T = 4 };

template <int EPI>
__global__ __launch_bounds__(256) void gemm_kernel(const f16* __restrict__ A,
                                                   const f16* __restrict__ W,
                                                   const float* __restrict__ bias,
                                                   void* __restrict__ outv,
                                                   const float* __restrict__ res,
                                                   int M, int N, int K) {
  __shared__ __align__(128) f16 bbuf[2][64 * 64];
  const int warp = threadIdx.x >> 5;
  const int lane = threadIdx.x & 31;
  const int hi = lane >> 4, ln = lane & 15;
  const int nStrips = N >> 6;
  const int row0 = (int)(blockIdx.x / nStrips) * 128 + warp * 16;
  const int col0 = (int)(blockIdx.x % nStrips) * 64;
  const int srow   = threadIdx.x >> 2;   // 0..63: weight row (output col) staged
  const int schunk = threadIdx.x & 3;    // 32-byte chunk of the 64-half slice

  // prologue: stage K-slab 0 (64 cols x 64 k = 8KB) into bbuf[0]
  {
    const f16* src = W + (size_t)(col0 + srow) * K + schunk * 16;
    f16* dst = &bbuf[0][srow * 64 + schunk * 16];
    async_copy_b128(src, dst);
    async_copy_b128(src + 8, dst + 8);
  }

  v8f acc[4] = {};
  const f16* Arow = A + (size_t)row0 * K;
  int cur = 0;
  for (int k0 = 0; k0 < K; k0 += 64) {
    wait_async0();       // this wave's staging of bbuf[cur] has landed
    __syncthreads();     // every wave's staging landed; prior readers drained
    if (k0 + 64 < K) {   // overlap: stage next K-slab into the other buffer
      const f16* src = W + (size_t)(col0 + srow) * K + (k0 + 64) + schunk * 16;
      f16* dst = &bbuf[cur ^ 1][srow * 64 + schunk * 16];
      async_copy_b128(src, dst);
      async_copy_b128(src + 8, dst + 8);
    }
    __builtin_prefetch(Arow + (size_t)ln * K + k0 + 128, 0, 0);
#pragma unroll
    for (int kk = 0; kk < 64; kk += 32) {
      const v16h a = load_a(Arow, K, k0 + kk, lane);
#pragma unroll
      for (int t = 0; t < 4; ++t) {
        const v16h b = load_bt(&bbuf[cur][(t * 16) * 64], 64, kk, lane);
        acc[t] = wmma_f16(a, b, acc[t]);
      }
    }
    cur ^= 1;
  }
#pragma unroll
  for (int t = 0; t < 4; ++t) {
#pragma unroll
    for (int r = 0; r < 8; ++r) {
      const int m = row0 + r + hi * 8;
      const int n = col0 + t * 16 + ln;
      float val = acc[t][r] + bias[n];
      if (EPI == EPI_F16_RELU) val = fmaxf(val, 0.0f);
      if (EPI == EPI_F32_RES) {
        ((float*)outv)[(size_t)m * N + n] = val + res[(size_t)m * N + n];
      } else if (EPI == EPI_F16_HEADS) {
        const int bb = m >> 11, s = m & (S_ - 1);     // m = b*S + s
        const int hh = n >> 6, dk = n & 63;           // n = h*DK + dk
        ((f16*)outv)[(((size_t)(bb * H_ + hh)) * S_ + s) * DK_ + dk] = (f16)val;
      } else if (EPI == EPI_F16_HEADS_T) {            // V stored [B,H,DK,S]
        const int bb = m >> 11, s = m & (S_ - 1);
        const int hh = n >> 6, dk = n & 63;
        ((f16*)outv)[(((size_t)(bb * H_ + hh)) * DK_ + dk) * S_ + s] = (f16)val;
      } else {
        ((f16*)outv)[(size_t)m * N + n] = (f16)val;
      }
    }
  }
}

// ---------------------------------------------------------------- flash attention
// q,k: [B,H,S,DK] f16.  v: [B,H,DK,S] f16 (transposed).  out: [B,S,D] f16.
// One wave = 16 query rows, streams 32 keys per iteration.

__global__ __launch_bounds__(128) void attn_kernel(const f16* __restrict__ q,
                                                   const f16* __restrict__ k,
                                                   const f16* __restrict__ v,
                                                   const int* __restrict__ mask,
                                                   f16* __restrict__ out) {
  __shared__ __align__(16) f16 pbuf[4][16 * 32];
  const int warp = threadIdx.x >> 5;
  const int lane = threadIdx.x & 31;
  const int hi = lane >> 4, ln = lane & 15;
  const int qBlocks = S_ / 64;
  const int bh = blockIdx.x / qBlocks;
  const int qb = blockIdx.x % qBlocks;
  const int b = bh / H_, h = bh % H_;
  const int qrow0 = qb * 64 + warp * 16;

  const f16* qp = q + (size_t)bh * S_ * DK_;
  const f16* kp = k + (size_t)bh * S_ * DK_;
  const f16* vp = v + (size_t)bh * DK_ * S_;   // [DK][S]
  const int* mk = mask + (size_t)b * S_;

  const v16h qa0 = load_a(qp + (size_t)qrow0 * DK_, DK_, 0, lane);
  const v16h qa1 = load_a(qp + (size_t)qrow0 * DK_, DK_, 32, lane);

  v8f o[4] = {};
  float mrow[8], lrow[8];
#pragma unroll
  for (int r = 0; r < 8; ++r) { mrow[r] = -__builtin_inff(); lrow[r] = 0.f; }

  for (int j = 0; j < S_; j += 32) {
    v8f s0 = {}, s1 = {};
    {
      v16h kb = load_bt(kp + (size_t)j * DK_, DK_, 0, lane);
      s0 = wmma_f16(qa0, kb, s0);
      kb = load_bt(kp + (size_t)j * DK_, DK_, 32, lane);
      s0 = wmma_f16(qa1, kb, s0);
      kb = load_bt(kp + (size_t)(j + 16) * DK_, DK_, 0, lane);
      s1 = wmma_f16(qa0, kb, s1);
      kb = load_bt(kp + (size_t)(j + 16) * DK_, DK_, 32, lane);
      s1 = wmma_f16(qa1, kb, s1);
    }
    const bool msk0 = (mk[j + ln] == 0);
    const bool msk1 = (mk[j + 16 + ln] == 0);
#pragma unroll
    for (int r = 0; r < 8; ++r) {
      float a0 = s0[r] * 0.125f; if (msk0) a0 = -1e9f;   // 1/sqrt(DK)=0.125
      float a1 = s1[r] * 0.125f; if (msk1) a1 = -1e9f;
      s0[r] = a0; s1[r] = a1;
    }
#pragma unroll
    for (int r = 0; r < 8; ++r) {
      float cm = fmaxf(s0[r], s1[r]);
#pragma unroll
      for (int off = 1; off < 16; off <<= 1) cm = fmaxf(cm, __shfl_xor(cm, off, 32));
      const float mn = fmaxf(mrow[r], cm);
      const float alpha = __expf(mrow[r] - mn);
      mrow[r] = mn;
      const float p0 = __expf(s0[r] - mn);
      const float p1 = __expf(s1[r] - mn);
      s0[r] = p0; s1[r] = p1;
      float rsum = p0 + p1;
#pragma unroll
      for (int off = 1; off < 16; off <<= 1) rsum += __shfl_xor(rsum, off, 32);
      lrow[r] = lrow[r] * alpha + rsum;
#pragma unroll
      for (int t = 0; t < 4; ++t) o[t][r] *= alpha;   // row r+8*hi scaling (C-layout)
    }
    __syncthreads();   // protect pbuf against previous iteration's readers
#pragma unroll
    for (int r = 0; r < 8; ++r) {
      const int m = r + hi * 8;
      pbuf[warp][m * 32 + ln]      = (f16)s0[r];
      pbuf[warp][m * 32 + 16 + ln] = (f16)s1[r];
    }
    __syncthreads();
    const v16h pa = load_a(&pbuf[warp][0], 32, 0, lane);
#pragma unroll
    for (int t = 0; t < 4; ++t) {
      // B[k][n] = Vt[(t*16+n)][j+k]: two 8-half runs per lane, 16 apart
      const v16h vb = load_bt(vp + (size_t)(t * 16) * S_ + j, S_, 0, lane);
      o[t] = wmma_f16(pa, vb, o[t]);
    }
  }
#pragma unroll
  for (int r = 0; r < 8; ++r) {
    const float inv = 1.0f / lrow[r];
    const int s = qrow0 + r + hi * 8;
#pragma unroll
    for (int t = 0; t < 4; ++t) {
      const int d = h * DK_ + t * 16 + ln;
      out[((size_t)(b * S_ + s)) * D_ + d] = (f16)(o[t][r] * inv);
    }
  }
}

// ---------------------------------------------------------------- host launcher

extern "C" void kernel_launch(void* const* d_in, const int* in_sizes, int n_in,
                              void* d_out, int out_size, void* d_ws, size_t ws_size,
                              hipStream_t stream) {
  (void)in_sizes; (void)n_in; (void)out_size; (void)ws_size;
  const float* x  = (const float*)d_in[0];
  const int*  msk = (const int*)d_in[1];
  const float* wq = (const float*)d_in[2];
  const float* bq = (const float*)d_in[3];
  const float* wk = (const float*)d_in[4];
  const float* bk = (const float*)d_in[5];
  const float* wv = (const float*)d_in[6];
  const float* bv = (const float*)d_in[7];
  const float* wo = (const float*)d_in[8];
  const float* bo = (const float*)d_in[9];
  const float* w1 = (const float*)d_in[10];
  const float* b1 = (const float*)d_in[11];
  const float* w2 = (const float*)d_in[12];
  const float* b2 = (const float*)d_in[13];
  const float* a1 = (const float*)d_in[14];
  const float* c1 = (const float*)d_in[15];
  const float* a2 = (const float*)d_in[16];
  const float* c2 = (const float*)d_in[17];
  float* out = (float*)d_out;

  char* ws = (char*)d_ws;
  size_t off = 0;
  auto alloc = [&](size_t bytes) -> void* {
    void* p = ws + off;
    off = (off + bytes + 255) & ~(size_t)255;
    return p;
  };
  f16* xn    = (f16*)alloc((size_t)M_ * D_ * 2);        // LN1 output
  f16* wq16  = (f16*)alloc((size_t)D_ * D_ * 2);
  f16* wk16  = (f16*)alloc((size_t)D_ * D_ * 2);
  f16* wv16  = (f16*)alloc((size_t)D_ * D_ * 2);
  f16* wo16  = (f16*)alloc((size_t)D_ * D_ * 2);
  f16* w1_16 = (f16*)alloc((size_t)DFF_ * D_ * 2);
  f16* w2_16 = (f16*)alloc((size_t)D_ * DFF_ * 2);
  f16* q16   = (f16*)alloc((size_t)M_ * D_ * 2);        // [B,H,S,DK]
  f16* k16   = (f16*)alloc((size_t)M_ * D_ * 2);        // [B,H,S,DK]
  f16* v16   = (f16*)alloc((size_t)M_ * D_ * 2);        // [B,H,DK,S] (transposed)
  f16* at16  = (f16*)alloc((size_t)M_ * D_ * 2);        // attention out [B,S,D]
  float* hbuf= (float*)alloc((size_t)M_ * D_ * 4);      // residual-1 output
  f16* hn    = (f16*)alloc((size_t)M_ * D_ * 2);        // LN2 output
  f16* ffh   = (f16*)alloc((size_t)M_ * DFF_ * 2);      // ReLU hidden

  // weight conversions
  f32_to_f16_kernel<<<(D_ * D_) / 256, 256, 0, stream>>>(wq, wq16, D_ * D_);
  f32_to_f16_kernel<<<(D_ * D_) / 256, 256, 0, stream>>>(wk, wk16, D_ * D_);
  f32_to_f16_kernel<<<(D_ * D_) / 256, 256, 0, stream>>>(wv, wv16, D_ * D_);
  f32_to_f16_kernel<<<(D_ * D_) / 256, 256, 0, stream>>>(wo, wo16, D_ * D_);
  f32_to_f16_kernel<<<(DFF_ * D_) / 256, 256, 0, stream>>>(w1, w1_16, DFF_ * D_);
  f32_to_f16_kernel<<<(DFF_ * D_) / 256, 256, 0, stream>>>(w2, w2_16, DFF_ * D_);

  // LN1
  ln_kernel<<<M_, 256, 0, stream>>>(x, a1, c1, xn);

  // Q,K,V projections: block = 128 rows x 64 cols
  const int gProj = (M_ / 128) * (D_ / 64);             // 1024
  gemm_kernel<EPI_F16_HEADS><<<gProj, 256, 0, stream>>>(xn, wq16, bq, q16, nullptr, M_, D_, D_);
  gemm_kernel<EPI_F16_HEADS><<<gProj, 256, 0, stream>>>(xn, wk16, bk, k16, nullptr, M_, D_, D_);
  gemm_kernel<EPI_F16_HEADS_T><<<gProj, 256, 0, stream>>>(xn, wv16, bv, v16, nullptr, M_, D_, D_);

  // attention (4 waves/block, 64 query rows per block)
  attn_kernel<<<B_ * H_ * (S_ / 64), 128, 0, stream>>>(q16, k16, v16, msk, at16);

  // O projection + residual-1: h = x + attn @ wo^T + bo
  gemm_kernel<EPI_F32_RES><<<gProj, 256, 0, stream>>>(at16, wo16, bo, hbuf, x, M_, D_, D_);

  // LN2
  ln_kernel<<<M_, 256, 0, stream>>>(hbuf, a2, c2, hn);

  // FFN1 + ReLU
  const int gF1 = (M_ / 128) * (DFF_ / 64);             // 4096
  gemm_kernel<EPI_F16_RELU><<<gF1, 256, 0, stream>>>(hn, w1_16, b1, ffh, nullptr, M_, DFF_, D_);

  // FFN2 + residual-2 -> final output
  gemm_kernel<EPI_F32_RES><<<gProj, 256, 0, stream>>>(ffh, w2_16, b2, out, hbuf, M_, D_, DFF_);
}
